// GAT_LSTM_3667902070977
// MI455X (gfx1250) — matrix operand order repaired and checked by reading, not compiled, not run on hardware
//
#include <hip/hip_runtime.h>
#include <stdint.h>

#define N_NODESC 20000
#define N_EDGESC 320000
#define N_E2C    340000     // edges + self loops
#define BATCHC   4096
#define SEQ_LENC 50
#define SEQ_FC   32
#define NODE_FC  128
#define EDGE_DIMC 16
#define GAT_OUTC 64
#define HEADSC   4
#define LSTM_HC  128

typedef __attribute__((ext_vector_type(16))) __bf16 v16bf;
typedef __attribute__((ext_vector_type(8)))  float  v8f;

union BF16Frag { uint4 q[2]; v16bf v; };

__device__ __forceinline__ unsigned short f2bf(float x) {
  unsigned u = __float_as_uint(x);
  unsigned r = u + 0x7fffu + ((u >> 16) & 1u);   // round-to-nearest-even
  return (unsigned short)(r >> 16);
}

__device__ __forceinline__ void atomicMaxF(float* a, float v) {
  if (v >= 0.f) atomicMax((int*)a, __float_as_int(v));
  else          atomicMin((unsigned int*)a, __float_as_uint(v));
}

__device__ __forceinline__ float sigmoidf_(float x) { return 1.f / (1.f + __expf(-x)); }

// ---------------------------------------------------------------- utilities
__global__ void k_fill(float* p, float v, size_t n) {
  size_t i  = (size_t)blockIdx.x * blockDim.x + threadIdx.x;
  size_t st = (size_t)gridDim.x * blockDim.x;
  for (; i < n; i += st) p[i] = v;
}

__global__ void k_cast_bf16(const float* in, unsigned short* out, size_t n) {
  size_t i = (size_t)blockIdx.x * blockDim.x + threadIdx.x;
  if (i < n) out[i] = f2bf(in[i]);
}

// in: K x N row-major  ->  out: N x K row-major (bf16)
__global__ void k_transpose_cast(const float* in, unsigned short* out, int K, int N) {
  int i = blockIdx.x * blockDim.x + threadIdx.x;
  if (i >= K * N) return;
  int n = i / K, k = i % K;
  out[i] = f2bf(in[k * N + n]);
}

// ------------------------------------------------ edge transform + seg mean
__global__ void k_edge_te(const float* eattr, const float* w, const float* bias,
                          const int* ei, float* te2, float* sacc, float* cnt) {
  size_t i = (size_t)blockIdx.x * blockDim.x + threadIdx.x;
  if (i >= (size_t)N_EDGESC * GAT_OUTC) return;
  int e = (int)(i >> 6), n = (int)(i & 63);
  float s = bias[n];
  #pragma unroll
  for (int k = 0; k < EDGE_DIMC; ++k) s += eattr[(size_t)e * EDGE_DIMC + k] * w[k * GAT_OUTC + n];
  te2[i] = s;
  int d = ei[N_EDGESC + e];
  atomicAdd(&sacc[(size_t)d * GAT_OUTC + n], s);
  if (n == 0) atomicAdd(&cnt[d], 1.f);
}

__global__ void k_self_loop(const float* sacc, const float* cnt, float* te2) {
  size_t i = (size_t)blockIdx.x * blockDim.x + threadIdx.x;
  if (i >= (size_t)N_NODESC * GAT_OUTC) return;
  int n = (int)(i >> 6);
  float c = cnt[n]; c = c > 1.f ? c : 1.f;
  te2[(size_t)(N_EDGESC + n) * GAT_OUTC + (i & 63)] = sacc[i] / c;
}

// --------------------------------------------------------- WMMA GEMM (bf16)
// C[M,N](f32) = A[M,K](bf16 row-major) * Wt[N,K](bf16 row-major)^T
// Block tile 32x256; 8 waves: wave&1 -> m-tile, wave>>1 -> group of 4 n-tiles.
// Per K-chunk: A loaded once, all 4 B fragments batch-loaded before the WMMA
// burst so loads overlap matrix execution (partial loadcnt waits).
// grid (N/256, M/32). Requires M%32==0, N%256==0, K%32==0.
__global__ void k_wmma_gemm(const unsigned short* __restrict__ A,
                            const unsigned short* __restrict__ Wt,
                            float* __restrict__ C, int M, int N, int K) {
  int wave = threadIdx.x >> 5, lane = threadIdx.x & 31;
  int mt  = (blockIdx.y << 1) + (wave & 1);
  int nt0 = blockIdx.x * 16 + (wave >> 1) * 4;
  int row = mt * 16 + (lane & 15);
  int col0 = nt0 * 16 + (lane & 15);
  int ka = (lane < 16) ? 0 : 16;   // A: 16 contiguous K
  int kb = (lane < 16) ? 0 : 16;   // B: 16 contiguous K
  v8f acc[4] = {};
  for (int kc = 0; kc < K; kc += 32) {
    BF16Frag a, b[4];
    const unsigned short* ap = A + (size_t)row * K + kc + ka;
    a.q[0] = *(const uint4*)(ap);
    a.q[1] = *(const uint4*)(ap + 8);
    #pragma unroll
    for (int j = 0; j < 4; ++j) {
      const unsigned short* bp = Wt + (size_t)(col0 + j * 16) * K + kc + kb;
      b[j].q[0] = *(const uint4*)(bp);
      b[j].q[1] = *(const uint4*)(bp + 8);
    }
    #pragma unroll
    for (int j = 0; j < 4; ++j)
      acc[j] = __builtin_amdgcn_wmma_f32_16x16x32_bf16(false, a.v, false, b[j].v, (short)0, acc[j], false, false);
  }
  int mbase = mt * 16 + ((lane < 16) ? 0 : 8);
  #pragma unroll
  for (int j = 0; j < 4; ++j)
    #pragma unroll
    for (int i = 0; i < 8; ++i)
      C[(size_t)(mbase + i) * N + col0 + j * 16] = acc[j][i];
}

// --------------------------------------- effective attention weight folding
__global__ void k_eff(const float* W, const float* att, float* eff, int K) {
  int i = blockIdx.x * blockDim.x + threadIdx.x;
  if (i >= K * HEADSC) return;
  int k = i >> 2, h = i & 3;
  float s = 0.f;
  #pragma unroll
  for (int c = 0; c < GAT_OUTC; ++c) s += W[(size_t)k * (HEADSC * GAT_OUTC) + h * GAT_OUTC + c] * att[h * GAT_OUTC + c];
  eff[i] = s;
}

__global__ void k_node_att(const float* x, const float* effs, const float* effd,
                           float* as_, float* ad_) {
  int i = blockIdx.x * blockDim.x + threadIdx.x;
  if (i >= N_NODESC * HEADSC) return;
  int n = i >> 2, h = i & 3;
  float ss = 0.f, sd = 0.f;
  for (int k = 0; k < NODE_FC; ++k) {
    float xv = x[(size_t)n * NODE_FC + k];
    ss += xv * effs[k * 4 + h];
    sd += xv * effd[k * 4 + h];
  }
  as_[i] = ss; ad_[i] = sd;
}

__global__ void k_edge_att(const float* te2, const float* weff, float* ae) {
  int i = blockIdx.x * blockDim.x + threadIdx.x;
  if (i >= N_E2C * HEADSC) return;
  int e = i >> 2, h = i & 3;
  float s = 0.f;
  for (int k = 0; k < GAT_OUTC; ++k) s += te2[(size_t)e * GAT_OUTC + k] * weff[k * 4 + h];
  ae[i] = s;
}

__global__ void k_logits(const int* ei, const float* as_, const float* ad_,
                         float* ae, float* m) {
  int i = blockIdx.x * blockDim.x + threadIdx.x;
  if (i >= N_E2C * HEADSC) return;
  int e = i >> 2, h = i & 3;
  int s = (e < N_EDGESC) ? ei[e] : (e - N_EDGESC);
  int d = (e < N_EDGESC) ? ei[N_EDGESC + e] : (e - N_EDGESC);
  float l = as_[s * 4 + h] + ad_[d * 4 + h] + ae[i];
  l = (l > 0.f) ? l : 0.2f * l;
  ae[i] = l;
  atomicMaxF(&m[d * 4 + h], l);
}

__global__ void k_expsum(const int* ei, float* ae, const float* m, float* den) {
  int i = blockIdx.x * blockDim.x + threadIdx.x;
  if (i >= N_E2C * HEADSC) return;
  int e = i >> 2, h = i & 3;
  int d = (e < N_EDGESC) ? ei[N_EDGESC + e] : (e - N_EDGESC);
  float p = __expf(ae[i] - m[d * 4 + h]);
  ae[i] = p;
  atomicAdd(&den[d * 4 + h], p);
}

__global__ void k_scatter(const int* ei, const float* p, const float* den,
                          const float* hm, float* o) {
  int e = blockIdx.x;
  int h = threadIdx.x >> 6, c = threadIdx.x & 63;
  int s = (e < N_EDGESC) ? ei[e] : (e - N_EDGESC);
  int d = (e < N_EDGESC) ? ei[N_EDGESC + e] : (e - N_EDGESC);
  float alpha = p[e * 4 + h] / den[d * 4 + h];
  atomicAdd(&o[(size_t)d * GAT_OUTC + c],
            0.25f * alpha * hm[(size_t)s * (HEADSC * GAT_OUTC) + h * GAT_OUTC + c]);
}

// ------------------------------------------- build LSTM layer-0 input (bf16)
__global__ void k_build_x0(const float* seq, const int* nidx,
                           const float* o1, const float* b1,
                           const float* o2, const float* b2,
                           unsigned short* X0) {
  size_t i = (size_t)blockIdx.x * blockDim.x + threadIdx.x;
  const int IN = SEQ_FC + 2 * GAT_OUTC;  // 160
  if (i >= (size_t)BATCHC * SEQ_LENC * IN) return;
  int k = (int)(i % IN);
  size_t bt = i / IN;
  int t = (int)(bt % SEQ_LENC);
  int b = (int)(bt / SEQ_LENC);
  float v;
  if (k < SEQ_FC) v = seq[((size_t)b * SEQ_LENC + t) * SEQ_FC + k];
  else if (k < SEQ_FC + GAT_OUTC) { int c = k - SEQ_FC;            v = o1[(size_t)nidx[b] * GAT_OUTC + c] + b1[c]; }
  else                            { int c = k - SEQ_FC - GAT_OUTC; v = o2[(size_t)nidx[b] * GAT_OUTC + c] + b2[c]; }
  X0[i] = f2bf(v);
}

// Wcat[g, 0:IN] = w_ih[g,:]; Wcat[g, IN:IN+128] = w_hh[g,:]; bcat = b_ih+b_hh
__global__ void k_prep_lstm(const float* wih, const float* whh,
                            const float* bih, const float* bhh,
                            unsigned short* Wc, float* bc, int IN) {
  int i = blockIdx.x * blockDim.x + threadIdx.x;
  int KD = IN + LSTM_HC;
  if (i < 4 * LSTM_HC) bc[i] = bih[i] + bhh[i];
  if (i >= 4 * LSTM_HC * KD) return;
  int g = i / KD, k = i % KD;
  Wc[i] = f2bf(k < IN ? wih[(size_t)g * IN + k] : whh[(size_t)g * LSTM_HC + (k - IN)]);
}

// ------------------------------------------------- persistent LSTM layer
// One workgroup owns 32 batch rows across all 50 timesteps; h/c/gates in LDS.
// Per step: gates(32x512) = [x_t | h](bf16,LDS) @ Wcat^T via WMMA.
// Wave w: m-tile = w>>2 (constant per wave); per K-chunk the A fragment is
// loaded once and all 8 B fragments are batch-loaded before the WMMA burst.
extern __shared__ char lstm_smem[];

template <int IN>
__global__ void k_lstm(const unsigned short* __restrict__ Xin,
                       const unsigned short* __restrict__ Wcat,
                       const float* __restrict__ bcat,
                       unsigned short* __restrict__ Hout,   // bf16 seq out (layer0) or null
                       float* __restrict__ hfinal) {        // last h (layer1) or null
  const int KD = IN + LSTM_HC;                 // 288 or 256
  unsigned short* xh = (unsigned short*)lstm_smem;           // 32*KD bf16
  float* gates = (float*)(lstm_smem + 32 * KD * 2);          // 32*512
  float* hbuf  = gates + 32 * 512;                           // 32*128
  float* cbuf  = hbuf + 32 * LSTM_HC;                        // 32*128
  int tid = threadIdx.x, wave = tid >> 5, lane = tid & 31;
  int b0 = blockIdx.x * 32;

  // pull fused weights toward this WGP's caches (global_prefetch_b8)
  for (size_t o = (size_t)tid * 64; o < (size_t)512 * KD * 2; o += 256 * 64)
    __builtin_prefetch((const char*)Wcat + o, 0, 1);

  for (int i = tid; i < 32 * LSTM_HC; i += 256) { hbuf[i] = 0.f; cbuf[i] = 0.f; }
  __syncthreads();

  const int mt  = wave >> 2;               // constant per wave: 0 or 1
  const int g0  = (wave & 3) * 128 + (lane & 15);  // first of 8 gate columns
  const int row = mt * 16 + (lane & 15);
  const int ka  = (lane < 16) ? 0 : 16;
  const int kb  = (lane < 16) ? 0 : 16;
  const int mbase = mt * 16 + ((lane < 16) ? 0 : 8);

  for (int t = 0; t < SEQ_LENC; ++t) {
    // stage [x_t | h] as bf16 into LDS
    for (int i = tid; i < 32 * IN; i += 256) {
      int r = i / IN, k = i % IN;
      xh[r * KD + k] = Xin[((size_t)(b0 + r) * SEQ_LENC + t) * IN + k];
    }
    for (int i = tid; i < 32 * LSTM_HC; i += 256) {
      int r = i >> 7, k = i & 127;
      xh[r * KD + IN + k] = f2bf(hbuf[i]);
    }
    __syncthreads();

    v8f acc[8];
    #pragma unroll
    for (int j = 0; j < 8; ++j) acc[j] = v8f{};
    for (int kc = 0; kc < KD; kc += 32) {
      BF16Frag a, b[8];
      const unsigned short* ap = xh + row * KD + kc + ka;
      a.q[0] = *(const uint4*)(ap);
      a.q[1] = *(const uint4*)(ap + 8);
      #pragma unroll
      for (int j = 0; j < 8; ++j) {
        const unsigned short* bp = Wcat + (size_t)(g0 + j * 16) * KD + kc + kb;
        b[j].q[0] = *(const uint4*)(bp);
        b[j].q[1] = *(const uint4*)(bp + 8);
      }
      #pragma unroll
      for (int j = 0; j < 8; ++j)
        acc[j] = __builtin_amdgcn_wmma_f32_16x16x32_bf16(false, a.v, false, b[j].v, (short)0, acc[j], false, false);
    }
    #pragma unroll
    for (int j = 0; j < 8; ++j) {
      int g = g0 + j * 16;
      float bias = bcat[g];
      #pragma unroll
      for (int i = 0; i < 8; ++i) gates[(mbase + i) * 512 + g] = acc[j][i] + bias;
    }
    __syncthreads();

    // elementwise LSTM cell (gate order i,f,g,o)
    for (int i = tid; i < 32 * LSTM_HC; i += 256) {
      int r = i >> 7, k = i & 127;
      float gi = sigmoidf_(gates[r * 512 + k]);
      float gf = sigmoidf_(gates[r * 512 + 128 + k]);
      float gg = tanhf(gates[r * 512 + 256 + k]);
      float go = sigmoidf_(gates[r * 512 + 384 + k]);
      float c = gf * cbuf[i] + gi * gg;
      float h = go * tanhf(c);
      cbuf[i] = c; hbuf[i] = h;
      if (Hout)   Hout[((size_t)(b0 + r) * SEQ_LENC + t) * LSTM_HC + k] = f2bf(h);
      if (hfinal && t == SEQ_LENC - 1) hfinal[(size_t)(b0 + r) * LSTM_HC + k] = h;
    }
    __syncthreads();
  }
}

__global__ void k_fc(const float* hfin, const float* fcw, const float* fcb, float* out) {
  int b = blockIdx.x * blockDim.x + threadIdx.x;
  if (b >= BATCHC) return;
  float s = fcb[0];
  for (int k = 0; k < LSTM_HC; ++k) s += hfin[(size_t)b * LSTM_HC + k] * fcw[k];
  out[b] = s;
}

// ---------------------------------------------------------------- launcher
extern "C" void kernel_launch(void* const* d_in, const int* in_sizes, int n_in,
                              void* d_out, int out_size, void* d_ws, size_t ws_size,
                              hipStream_t stream) {
  (void)in_sizes; (void)n_in; (void)out_size; (void)ws_size;
  const float* seq    = (const float*)d_in[0];
  const int*   ei     = (const int*)  d_in[1];
  const float* eattr  = (const float*)d_in[2];
  const float* nodef  = (const float*)d_in[3];
  const int*   nidx   = (const int*)  d_in[4];
  const float* eat_w  = (const float*)d_in[5];
  const float* eat_b  = (const float*)d_in[6];
  const float* g_lin_w[2]   = { (const float*)d_in[7],  (const float*)d_in[13] };
  const float* g_att_src[2] = { (const float*)d_in[8],  (const float*)d_in[14] };
  const float* g_att_dst[2] = { (const float*)d_in[9],  (const float*)d_in[15] };
  const float* g_att_edge[2]= { (const float*)d_in[10], (const float*)d_in[16] };
  const float* g_lin_ew[2]  = { (const float*)d_in[11], (const float*)d_in[17] };
  const float* g_bias[2]    = { (const float*)d_in[12], (const float*)d_in[18] };
  const float* w_ih0 = (const float*)d_in[19];
  const float* w_hh0 = (const float*)d_in[20];
  const float* b_ih0 = (const float*)d_in[21];
  const float* b_hh0 = (const float*)d_in[22];
  const float* w_ih1 = (const float*)d_in[23];
  const float* w_hh1 = (const float*)d_in[24];
  const float* b_ih1 = (const float*)d_in[25];
  const float* b_hh1 = (const float*)d_in[26];
  const float* fc_w  = (const float*)d_in[27];
  const float* fc_b  = (const float*)d_in[28];
  float* out = (float*)d_out;

  float* ws = (float*)d_ws;
  size_t off = 0;
  auto A = [&](size_t n) { size_t r = off; off += (n + 3) & ~(size_t)3; return r; };
  size_t te2_o  = A((size_t)N_E2C * GAT_OUTC);
  size_t sacc_o = A((size_t)N_NODESC * GAT_OUTC);
  size_t cnt_o  = A(N_NODESC);
  size_t nfbf_o = A((size_t)N_NODESC * NODE_FC / 2);           // u16
  size_t linT_o[2] = { A(256 * 128 / 2), A(256 * 128 / 2) };   // u16
  size_t hm_o[2]   = { A((size_t)N_NODESC * 256), A((size_t)N_NODESC * 256) };
  size_t effs_o[2] = { A(128 * 4), A(128 * 4) };
  size_t effd_o[2] = { A(128 * 4), A(128 * 4) };
  size_t weff_o[2] = { A(64 * 4),  A(64 * 4) };
  size_t as_o[2]   = { A((size_t)N_NODESC * 4), A((size_t)N_NODESC * 4) };
  size_t ad_o[2]   = { A((size_t)N_NODESC * 4), A((size_t)N_NODESC * 4) };
  size_t ae_o[2]   = { A((size_t)N_E2C * 4), A((size_t)N_E2C * 4) };
  size_t m_o[2]    = { A((size_t)N_NODESC * 4), A((size_t)N_NODESC * 4) };
  size_t dn_o[2]   = { A((size_t)N_NODESC * 4), A((size_t)N_NODESC * 4) };
  size_t o_o[2]    = { A((size_t)N_NODESC * GAT_OUTC), A((size_t)N_NODESC * GAT_OUTC) };
  size_t X0_o  = A((size_t)BATCHC * SEQ_LENC * 160 / 2);       // u16
  size_t X1_o  = A((size_t)BATCHC * SEQ_LENC * 128 / 2);       // u16
  size_t Wc0_o = A(512 * 288 / 2);                              // u16
  size_t Wc1_o = A(512 * 256 / 2);                              // u16
  size_t bc0_o = A(512), bc1_o = A(512);
  size_t hfin_o = A((size_t)BATCHC * LSTM_HC);

  const int B = 256;
  auto nb = [](size_t n) { return (unsigned)((n + 255) / 256); };

  // zero / init accumulators
  k_fill<<<4096, B, 0, stream>>>(ws + sacc_o, 0.f, (size_t)N_NODESC * GAT_OUTC);
  k_fill<<<128,  B, 0, stream>>>(ws + cnt_o,  0.f, (size_t)N_NODESC);
  for (int br = 0; br < 2; ++br) {
    k_fill<<<512, B, 0, stream>>>(ws + m_o[br],  -3.0e38f, (size_t)N_NODESC * 4);
    k_fill<<<512, B, 0, stream>>>(ws + dn_o[br], 0.f, (size_t)N_NODESC * 4);
    k_fill<<<4096, B, 0, stream>>>(ws + o_o[br], 0.f, (size_t)N_NODESC * GAT_OUTC);
  }

  // edge transform + self-loop mean
  k_edge_te<<<nb((size_t)N_EDGESC * GAT_OUTC), B, 0, stream>>>(
      eattr, eat_w, eat_b, ei, ws + te2_o, ws + sacc_o, ws + cnt_o);
  k_self_loop<<<nb((size_t)N_NODESC * GAT_OUTC), B, 0, stream>>>(
      ws + sacc_o, ws + cnt_o, ws + te2_o);

  // bf16 casts / transposes for WMMA
  k_cast_bf16<<<nb((size_t)N_NODESC * NODE_FC), B, 0, stream>>>(
      nodef, (unsigned short*)(ws + nfbf_o), (size_t)N_NODESC * NODE_FC);
  for (int br = 0; br < 2; ++br)
    k_transpose_cast<<<nb(128 * 256), B, 0, stream>>>(
        g_lin_w[br], (unsigned short*)(ws + linT_o[br]), 128, 256);

  for (int br = 0; br < 2; ++br) {
    // h = x @ lin_w  (20000x128 @ 128x256) via WMMA bf16
    dim3 grid(256 / 256, N_NODESC / 32);
    k_wmma_gemm<<<grid, B, 0, stream>>>(
        (const unsigned short*)(ws + nfbf_o), (const unsigned short*)(ws + linT_o[br]),
        ws + hm_o[br], N_NODESC, 256, 128);
    // folded attention weights
    k_eff<<<nb(128 * 4), B, 0, stream>>>(g_lin_w[br], g_att_src[br], ws + effs_o[br], 128);
    k_eff<<<nb(128 * 4), B, 0, stream>>>(g_lin_w[br], g_att_dst[br], ws + effd_o[br], 128);
    k_eff<<<nb(64 * 4),  B, 0, stream>>>(g_lin_ew[br], g_att_edge[br], ws + weff_o[br], 64);
    k_node_att<<<nb((size_t)N_NODESC * 4), B, 0, stream>>>(
        nodef, ws + effs_o[br], ws + effd_o[br], ws + as_o[br], ws + ad_o[br]);
    k_edge_att<<<nb((size_t)N_E2C * 4), B, 0, stream>>>(ws + te2_o, ws + weff_o[br], ws + ae_o[br]);
    // softmax over incoming edges + scatter aggregate
    k_logits<<<nb((size_t)N_E2C * 4), B, 0, stream>>>(
        ei, ws + as_o[br], ws + ad_o[br], ws + ae_o[br], ws + m_o[br]);
    k_expsum<<<nb((size_t)N_E2C * 4), B, 0, stream>>>(ei, ws + ae_o[br], ws + m_o[br], ws + dn_o[br]);
    k_scatter<<<N_E2C, B, 0, stream>>>(ei, ws + ae_o[br], ws + dn_o[br], ws + hm_o[br], ws + o_o[br]);
  }

  // LSTM input assembly + fused weights
  k_build_x0<<<nb((size_t)BATCHC * SEQ_LENC * 160), B, 0, stream>>>(
      seq, nidx, ws + o_o[0], g_bias[0], ws + o_o[1], g_bias[1],
      (unsigned short*)(ws + X0_o));
  k_prep_lstm<<<nb(512 * 288), B, 0, stream>>>(
      w_ih0, w_hh0, b_ih0, b_hh0, (unsigned short*)(ws + Wc0_o), ws + bc0_o, 160);
  k_prep_lstm<<<nb(512 * 256), B, 0, stream>>>(
      w_ih1, w_hh1, b_ih1, b_hh1, (unsigned short*)(ws + Wc1_o), ws + bc1_o, 128);

  // persistent WMMA LSTM layers (32 batch rows per workgroup, h/c/gates in LDS)
  size_t smem0 = 32 * (160 + 128) * 2 + 32 * 512 * 4 + 2 * 32 * 128 * 4;  // 116736 B
  size_t smem1 = 32 * (128 + 128) * 2 + 32 * 512 * 4 + 2 * 32 * 128 * 4;  // 114688 B
  k_lstm<160><<<BATCHC / 32, B, smem0, stream>>>(
      (const unsigned short*)(ws + X0_o), (const unsigned short*)(ws + Wc0_o),
      ws + bc0_o, (unsigned short*)(ws + X1_o), nullptr);
  k_lstm<128><<<BATCHC / 32, B, smem1, stream>>>(
      (const unsigned short*)(ws + X1_o), (const unsigned short*)(ws + Wc1_o),
      ws + bc1_o, nullptr, ws + hfin_o);

  k_fc<<<nb(BATCHC), B, 0, stream>>>(ws + hfin_o, fc_w, fc_b, out);
}
